// CAttention_3032246911403
// MI455X (gfx1250) — matrix operand order repaired and checked by reading, not compiled, not run on hardware
//
#include <hip/hip_runtime.h>

typedef __attribute__((ext_vector_type(16))) _Float16 v16h;
typedef __attribute__((ext_vector_type(8)))  _Float16 v8h;
typedef __attribute__((ext_vector_type(8)))  float    v8f;

#define BATCH   2
#define HEADS   8
#define NSEQ    4096
#define DHEAD   32
#define CH      256           // HEADS * DHEAD
#define WAVES   8             // waves per block (256 threads, wave32)
#define KCHUNK  64            // kv rows staged per iteration (2 x 32 sub-chunks)
#define LPADK   40            // ldsK row stride in halves (32 + 8 pad, 16B-aligned rows)
#define LPADV   72            // ldsV row stride in halves (64 + 8 pad, 16B-aligned rows)

// xor-16 lane swap via DS swizzle immediate (SWAPX16: xor=0x10, and=0x1f) — no index VGPR.
static __device__ __forceinline__ int   swz16 (int x)   { return __builtin_amdgcn_ds_swizzle(x, 0x401f); }
static __device__ __forceinline__ float swz16f(float x) { return __int_as_float(swz16(__float_as_int(x))); }
// pack two f32 -> one dword of two f16 (v_cvt_pk_f16_f32)
static __device__ __forceinline__ int pkh2(float a, float b) {
    auto t = __builtin_amdgcn_cvt_pkrtz(a, b);   // __fp16 ext_vector(2)
    return __builtin_bit_cast(int, t);
}

// Online-softmax update for one q-tile over a 32-kv sub-chunk.
// st0/st1: S^T tiles (C layout: lane = column q, rows kv = {8*lh+r, 16+8*lh+r}).
// Returns P^T B-operand fragment; rescales acc pair by alpha.
static __device__ __forceinline__ v16h softmax_update(const v8f st0, const v8f st1,
                                                      float& m_run, float& l_run,
                                                      v8f& acc0, v8f& acc1, int lh) {
    float tmax = fmaxf(st0[0], st0[1]);
    #pragma unroll
    for (int r = 2; r < 8; ++r) tmax = fmaxf(tmax, st0[r]);
    #pragma unroll
    for (int r = 0; r < 8; ++r) tmax = fmaxf(tmax, st1[r]);
    tmax = fmaxf(tmax, swz16f(tmax));                 // lanes L, L^16 share column q
    const float m_new = fmaxf(m_run, tmax);
    const float alpha = __builtin_amdgcn_exp2f(m_run - m_new);
    m_run = m_new;

    float p0[8], p1[8], psum = 0.0f;
    #pragma unroll
    for (int r = 0; r < 8; ++r) {
        p0[r] = __builtin_amdgcn_exp2f(st0[r] - m_new);
        p1[r] = __builtin_amdgcn_exp2f(st1[r] - m_new);
        psum += p0[r] + p1[r];
    }
    l_run = l_run * alpha + psum;
    #pragma unroll
    for (int i = 0; i < 8; ++i) { acc0[i] *= alpha; acc1[i] *= alpha; }

    // Pack to f16 dwords, then ONE xor-16 exchange per dword (4 total):
    // lane<16 needs partner's p0 (kv 8..15), lane>=16 needs partner's p1 (kv 16..23).
    int own0[4], own1[4];
    #pragma unroll
    for (int j = 0; j < 4; ++j) {
        own0[j] = pkh2(p0[2 * j], p0[2 * j + 1]);
        own1[j] = pkh2(p1[2 * j], p1[2 * j + 1]);
    }
    union { v16h h; int i[8]; } pf;
    #pragma unroll
    for (int j = 0; j < 4; ++j) {
        const int recv = swz16((lh == 0) ? own1[j] : own0[j]);
        pf.i[j]     = (lh == 0) ? own0[j] : recv;     // halves 0..7 : kv 0..7  / 16..23
        pf.i[4 + j] = (lh == 0) ? recv    : own1[j];  // halves 8..15: kv 8..15 / 24..31
    }
    return pf.h;
}

// Flash attention, S^T formulation, 2 q-tiles (32 q rows) per wave:
//   S^T = K(16kv x 32d) x Q^T(32d x 16q)    -> v_wmma_f32_16x16x32_f16 (K=d=32 exact)
//   O^T = V^T(16d x 32kv) x P^T(32kv x 16q) -> v_wmma_f32_16x16x32_f16 (accumulating)
// kf/vf A-operands are shared by both q-tiles -> 8 WMMAs per 32-kv sub-chunk.
__global__ __launch_bounds__(256) void fa_wmma_kernel(const float* __restrict__ q,
                                                      const float* __restrict__ k,
                                                      const float* __restrict__ v,
                                                      float* __restrict__ out) {
    __shared__ _Float16 ldsK[KCHUNK * LPADK];  // [kv][d]  row-major
    __shared__ _Float16 ldsV[DHEAD  * LPADV];  // [d][kv]  transposed

    const int tid  = threadIdx.x;
    const int lane = tid & 31;          // gfx1250 is wave32
    const int wave = tid >> 5;
    const int ln   = lane & 15;
    const int lh   = lane >> 4;

    const int bh    = blockIdx.x & (BATCH * HEADS - 1);
    const int b     = bh >> 3;
    const int h     = bh & 7;
    const int qBase = (blockIdx.x >> 4) * (WAVES * 32) + wave * 32;  // 32 q rows per wave

    // 1/sqrt(32) * log2(e): fold softmax scale + exp2 conversion into Q once.
    const float SCALE = 0.17677669529663687f * 1.4426950408889634f;

    // Q fragments (B operand): lane = column q, halves = d = lh*16 .. lh*16+15 (contiguous)
    v16h qfA, qfB;
    #pragma unroll
    for (int t = 0; t < 2; ++t) {
        const size_t qoff = ((size_t)b * NSEQ + (size_t)(qBase + t * 16 + ln)) * CH
                          + h * DHEAD + lh * 16;
        v16h f;
        #pragma unroll
        for (int i = 0; i < 4; ++i) {
            float4 x = *(const float4*)&q[qoff + 4 * i];
            f[4 * i + 0] = (_Float16)(x.x * SCALE);
            f[4 * i + 1] = (_Float16)(x.y * SCALE);
            f[4 * i + 2] = (_Float16)(x.z * SCALE);
            f[4 * i + 3] = (_Float16)(x.w * SCALE);
        }
        if (t == 0) qfA = f; else qfB = f;
    }

    v8f accA0 = {}, accA1 = {}, accB0 = {}, accB1 = {};
    float mA = -1e30f, lA = 0.0f, mB = -1e30f, lB = 0.0f;

    const size_t kvBase = (size_t)b * NSEQ * CH + h * DHEAD;

    // staging: 64 rows x 32 d = 2048 floats / 256 threads = 8 floats (one v8h K piece) each
    const int skv = tid >> 2;           // 0..63
    const int sd0 = (tid & 3) * 8;      // 0,8,16,24

    for (int kv0 = 0; kv0 < NSEQ; kv0 += KCHUNK) {
        __syncthreads();
        {
            const size_t goff = kvBase + (size_t)(kv0 + skv) * CH + sd0;
            const float4 k0 = *(const float4*)&k[goff];
            const float4 k1 = *(const float4*)&k[goff + 4];
            v8h kh = { (_Float16)k0.x, (_Float16)k0.y, (_Float16)k0.z, (_Float16)k0.w,
                       (_Float16)k1.x, (_Float16)k1.y, (_Float16)k1.z, (_Float16)k1.w };
            *(v8h*)&ldsK[skv * LPADK + sd0] = kh;
            const float4 v0 = *(const float4*)&v[goff];
            const float4 v1 = *(const float4*)&v[goff + 4];
            ldsV[(sd0 + 0) * LPADV + skv] = (_Float16)v0.x;
            ldsV[(sd0 + 1) * LPADV + skv] = (_Float16)v0.y;
            ldsV[(sd0 + 2) * LPADV + skv] = (_Float16)v0.z;
            ldsV[(sd0 + 3) * LPADV + skv] = (_Float16)v0.w;
            ldsV[(sd0 + 4) * LPADV + skv] = (_Float16)v1.x;
            ldsV[(sd0 + 5) * LPADV + skv] = (_Float16)v1.y;
            ldsV[(sd0 + 6) * LPADV + skv] = (_Float16)v1.z;
            ldsV[(sd0 + 7) * LPADV + skv] = (_Float16)v1.w;
            if (kv0 + KCHUNK < NSEQ) {  // global_prefetch_b8 of next chunk
                __builtin_prefetch(&k[goff + (size_t)KCHUNK * CH], 0, 1);
                __builtin_prefetch(&v[goff + (size_t)KCHUNK * CH], 0, 1);
            }
        }
        __syncthreads();

        #pragma unroll
        for (int c = 0; c < 2; ++c) {   // two 32-kv sub-chunks
            const int kvb = c * 32;
            const int dlo = lh * 16;    // A-operand K contiguous: {0..15}/{16..31}
            // K fragments (A operand, 16x32): lane = row kv
            v16h kf0, kf1;
            {
                v8h a0 = *(const v8h*)&ldsK[(kvb + ln)      * LPADK + dlo];
                v8h a1 = *(const v8h*)&ldsK[(kvb + ln)      * LPADK + 8 + dlo];
                v8h a2 = *(const v8h*)&ldsK[(kvb + 16 + ln) * LPADK + dlo];
                v8h a3 = *(const v8h*)&ldsK[(kvb + 16 + ln) * LPADK + 8 + dlo];
                #pragma unroll
                for (int i = 0; i < 8; ++i) { kf0[i] = a0[i]; kf0[8 + i] = a1[i];
                                              kf1[i] = a2[i]; kf1[8 + i] = a3[i]; }
            }
            const v8f z = {};
            v8f sA0 = __builtin_amdgcn_wmma_f32_16x16x32_f16(false, kf0, false, qfA, (short)0, z, false, false);
            v8f sA1 = __builtin_amdgcn_wmma_f32_16x16x32_f16(false, kf1, false, qfA, (short)0, z, false, false);
            v8f sB0 = __builtin_amdgcn_wmma_f32_16x16x32_f16(false, kf0, false, qfB, (short)0, z, false, false);
            v8f sB1 = __builtin_amdgcn_wmma_f32_16x16x32_f16(false, kf1, false, qfB, (short)0, z, false, false);

            // V^T fragments (A operand): issue DS loads early so latency hides under softmax
            const int klo = lh * 16;
            v8h b0 = *(const v8h*)&ldsV[(ln)      * LPADV + kvb + klo];
            v8h b1 = *(const v8h*)&ldsV[(ln)      * LPADV + kvb + 8 + klo];
            v8h b2 = *(const v8h*)&ldsV[(16 + ln) * LPADV + kvb + klo];
            v8h b3 = *(const v8h*)&ldsV[(16 + ln) * LPADV + kvb + 8 + klo];
            v16h vf0, vf1;
            #pragma unroll
            for (int i = 0; i < 8; ++i) { vf0[i] = b0[i]; vf0[8 + i] = b1[i];
                                          vf1[i] = b2[i]; vf1[8 + i] = b3[i]; }

            const v16h pA = softmax_update(sA0, sA1, mA, lA, accA0, accA1, lh);
            accA0 = __builtin_amdgcn_wmma_f32_16x16x32_f16(false, vf0, false, pA, (short)0, accA0, false, false);
            accA1 = __builtin_amdgcn_wmma_f32_16x16x32_f16(false, vf1, false, pA, (short)0, accA1, false, false);

            const v16h pB = softmax_update(sB0, sB1, mB, lB, accB0, accB1, lh);
            accB0 = __builtin_amdgcn_wmma_f32_16x16x32_f16(false, vf0, false, pB, (short)0, accB0, false, false);
            accB1 = __builtin_amdgcn_wmma_f32_16x16x32_f16(false, vf1, false, pB, (short)0, accB1, false, false);
        }
    }

    // ---- epilogue: combine l across lane halves, normalize, contiguous b128 stores ----
    #pragma unroll
    for (int t = 0; t < 2; ++t) {
        const float l_run = (t == 0) ? lA : lB;
        const v8f  a0 = (t == 0) ? accA0 : accB0;
        const v8f  a1 = (t == 0) ? accA1 : accB1;
        const float inv = 1.0f / (l_run + swz16f(l_run));
        // lane holds O^T column q; rows d = dh*16 + 8*lh + r  (r contiguous)
        const size_t ooff = ((size_t)b * NSEQ + (size_t)(qBase + t * 16 + ln)) * CH
                          + h * DHEAD + lh * 8;
        float4 w0 = { a0[0] * inv, a0[1] * inv, a0[2] * inv, a0[3] * inv };
        float4 w1 = { a0[4] * inv, a0[5] * inv, a0[6] * inv, a0[7] * inv };
        float4 w2 = { a1[0] * inv, a1[1] * inv, a1[2] * inv, a1[3] * inv };
        float4 w3 = { a1[4] * inv, a1[5] * inv, a1[6] * inv, a1[7] * inv };
        *(float4*)&out[ooff +  0] = w0;
        *(float4*)&out[ooff +  4] = w1;
        *(float4*)&out[ooff + 16] = w2;
        *(float4*)&out[ooff + 20] = w3;
    }
}

extern "C" void kernel_launch(void* const* d_in, const int* in_sizes, int n_in,
                              void* d_out, int out_size, void* d_ws, size_t ws_size,
                              hipStream_t stream) {
    const float* q = (const float*)d_in[0];
    const float* k = (const float*)d_in[1];
    const float* v = (const float*)d_in[2];
    float* out = (float*)d_out;
    const int qBlocksPerBH = NSEQ / (WAVES * 32);             // 16
    dim3 grid(qBlocksPerBH * BATCH * HEADS);                  // 256 blocks
    dim3 block(256);                                          // 8 waves (wave32)
    hipLaunchKernelGGL(fa_wmma_kernel, grid, block, 0, stream, q, k, v, out);
}
